// EarthSpecificBlock_38835094290617
// MI455X (gfx1250) — matrix operand order repaired and checked by reading, not compiled, not run on hardware
//
#include <hip/hip_runtime.h>
#include <hip/hip_bf16.h>
#include <cstdint>
#include <cstddef>

// ---------------- problem constants ----------------
#define DIM    256
#define HEADS  8
#define HD     32
#define NTOK   144      // tokens per window (2*6*12)
#define NW     40       // windows per batch-lon slice (nPl*nLat)
#define BWIN   20       // B_ = B*nLon
#define MTOT   115200   // B*L token rows
#define LTOK   57600    // L
#define HIDN   1024

typedef __attribute__((ext_vector_type(16))) __bf16        v16bf;
typedef __attribute__((ext_vector_type(8)))  float         v8f;
typedef __attribute__((ext_vector_type(4)))  unsigned int  v4u;
typedef __attribute__((ext_vector_type(8)))  int           v8i;
typedef __attribute__((ext_vector_type(4)))  int           v4i;

#if defined(__HIP_DEVICE_COMPILE__) && __has_builtin(__builtin_amdgcn_tensor_load_to_lds)
#define USE_TDM 1
#else
#define USE_TDM 0
#endif

union FragU { v4u q[2]; v16bf v; };

__device__ __forceinline__ unsigned short f2bf(float x) {
  union { float f; unsigned u; } a; a.f = x;
  unsigned u = a.u + 0x7FFFu + ((a.u >> 16) & 1u);   // RNE
  return (unsigned short)(u >> 16);
}
__device__ __forceinline__ float bf2f(unsigned short h) {
  union { unsigned u; float f; } a; a.u = ((unsigned)h) << 16;
  return a.f;
}

// ---- WMMA fragment loads from LDS (bf16, K=32 window, 64B rows unless noted) ----
// A 16x32 layout (ISA 7.12.2): lane<16 holds K {0..7,16..23}, lane>=16 holds {8..15,24..31}
__device__ __forceinline__ v16bf frag_a(const char* row_base, int lane_hi) {
  FragU f;
  f.q[0] = *(const v4u*)(row_base + 16 * lane_hi);
  f.q[1] = *(const v4u*)(row_base + 32 + 16 * lane_hi);
  return f.v;
}
// B 32x16 layout (n-major LDS rows): lane<16 holds K {0..7,16..23}, lane>=16 holds {8..15,24..31}
__device__ __forceinline__ v16bf frag_b(const char* row_base, int lane_hi) {
  FragU f;
  f.q[0] = *(const v4u*)(row_base + 16 * lane_hi);
  f.q[1] = *(const v4u*)(row_base + 32 + 16 * lane_hi);
  return f.v;
}
__device__ __forceinline__ v8f wmma_bf16(v16bf a, v16bf b, v8f c) {
  return __builtin_amdgcn_wmma_f32_16x16x32_bf16(false, a, false, b, (short)0, c,
                                                 false, false);
}

__device__ __forceinline__ void wait_tensor0() {
#if defined(__HIP_DEVICE_COMPILE__) && __has_builtin(__builtin_amdgcn_s_wait_tensorcnt)
  __builtin_amdgcn_s_wait_tensorcnt(0);
#endif
}

// ---- Tensor Data Mover: 2D bf16 tile (rows x 32 elems) global -> LDS ----
// D# per CDNA5 ISA ch.8: group0 = {count, lds_addr, global_addr, type=2},
// group1 = {data_size=2B, tensor_dim0=32, tensor_dim1=rows, tile 32 x rows,
//           tensor_dim0_stride = src leading dim}.  Tile rows land contiguously
// in LDS (64B per row), exactly the fragment layout the WMMA loop reads.
__device__ __forceinline__ void tdm_load_tile(unsigned lds_off,
                                              const unsigned short* gsrc,
                                              int rows, int src_ld) {
#if USE_TDM
  unsigned long long ga = (unsigned long long)(uintptr_t)gsrc;
  v4u g0;
  g0[0] = 1u;                                            // count=1, user mode
  g0[1] = lds_off;                                       // LDS byte address
  g0[2] = (unsigned)ga;                                  // global addr [31:0]
  g0[3] = ((unsigned)(ga >> 32) & 0x01FFFFFFu) | (2u << 30);  // [56:32] | type=2
  v8i g1;
  g1[0] = (int)(1u << 16);                               // data_size = 2 bytes
  g1[1] = (int)(32u << 16);                              // tensor_dim0 = 32
  g1[2] = (int)(((unsigned)rows & 0xFFFFu) << 16);       // tensor_dim1 = rows
  g1[3] = (int)(32u << 16);                              // tile_dim0 = 32
  g1[4] = (int)((unsigned)rows & 0xFFFFu);               // tile_dim1 = rows, dim2=0
  g1[5] = src_ld;                                        // dim0 stride lo32
  g1[6] = 0;
  g1[7] = 0;
  v4i z = {};
#if __clang_major__ >= 23
  v8i z8 = {};
  __builtin_amdgcn_tensor_load_to_lds(g0, g1, z, z, z8, 0);
#else
  __builtin_amdgcn_tensor_load_to_lds(g0, g1, z, z, 0);
#endif
#else
  (void)lds_off; (void)gsrc; (void)rows; (void)src_ld;
#endif
}

// ---- window <-> token index maps (RES 8x60x120, WIN 2x6x12) ----
__device__ __forceinline__ size_t tok2win(int t) {
  int b = t / LTOK, r = t % LTOK;
  int pl = r / 7200, r2 = r % 7200;
  int lat = r2 / 120, lon = r2 % 120;
  int bw = b * 10 + lon / 12;
  int w  = (pl >> 1) * 10 + lat / 6;
  int n  = (pl & 1) * 72 + (lat % 6) * 12 + (lon % 12);
  return (size_t)(bw * NW + w) * NTOK + n;
}
__device__ __forceinline__ size_t win2tok(int m) {
  int bw = m / (NW * NTOK), rem = m % (NW * NTOK);
  int w = rem / NTOK, n = rem % NTOK;
  int b = bw / 10, nlon = bw % 10;
  int pl  = (w / 10) * 2 + n / 72;
  int lat = (w % 10) * 6 + (n / 12) % 6;
  int lon = nlon * 12 + n % 12;
  return (size_t)b * LTOK + (size_t)(pl * 60 + lat) * 120 + lon;
}

// earth position index (closed form of get_earth_position_index for WIN=(2,6,12))
__device__ __forceinline__ int posidx(int n, int m) {
  int d0 = (n / 72) + 2 * (m / 72);
  int d1 = ((n / 12) % 6) + 6 * ((m / 12) % 6);
  int d2 = (n % 12) - (m % 12) + 11;
  return d0 * 828 + d1 * 23 + d2;   // 828 = (2*12-1)*6*6
}

// ---------------- fp32 -> bf16 convert (weights) ----------------
__global__ __launch_bounds__(256) void esb_cvt(const float* __restrict__ in,
                                               unsigned short* __restrict__ out, int n) {
  int i = blockIdx.x * 256 + threadIdx.x;
  if (i < n) out[i] = f2bf(in[i]);
}

// ---------------- LayerNorm (one wave per 256-wide row) ----------------
__global__ __launch_bounds__(256) void esb_ln(const float* __restrict__ x,
                                              const float* __restrict__ g,
                                              const float* __restrict__ b,
                                              unsigned short* __restrict__ out,
                                              int remap) {
  int row  = blockIdx.x * 8 + (threadIdx.x >> 5);
  int lane = threadIdx.x & 31;
  const float* xr = x + (size_t)row * DIM;
  float v[8], s = 0.f;
#pragma unroll
  for (int i = 0; i < 8; i++) { v[i] = xr[lane + 32 * i]; s += v[i]; }
#pragma unroll
  for (int o = 16; o; o >>= 1) s += __shfl_xor(s, o, 32);
  float mu = s * (1.f / DIM), var = 0.f;
#pragma unroll
  for (int i = 0; i < 8; i++) { float d = v[i] - mu; var += d * d; }
#pragma unroll
  for (int o = 16; o; o >>= 1) var += __shfl_xor(var, o, 32);
  float rs = rsqrtf(var * (1.f / DIM) + 1e-5f);
  size_t orow = remap ? tok2win(row) : (size_t)row;
  unsigned short* op = out + orow * DIM;
#pragma unroll
  for (int i = 0; i < 8; i++) {
    int c = lane + 32 * i;
    op[c] = f2bf((v[i] - mu) * rs * g[c] + b[c]);
  }
}

// ---------------- bf16 GEMM: Y[M,N] = A[M,K] * W[N,K]^T + bias ----------------
// MODE 0: store bf16        MODE 1: GELU, store bf16
// MODE 2: +res (win->token remap), store f32   MODE 3: +res (token), store f32
template <int MODE>
__global__ __launch_bounds__(256) void esb_gemm(const unsigned short* __restrict__ A,
                                                const unsigned short* __restrict__ W,
                                                const float* __restrict__ bias,
                                                int M, int N, int K,
                                                const float* __restrict__ res,
                                                float* __restrict__ outf,
                                                unsigned short* __restrict__ outh) {
  __shared__ __align__(16) char sA[2][64 * 64];    // 2 x (64 rows x 32 bf16)
  __shared__ __align__(16) char sB[2][128 * 64];   // 2 x (128 n-rows x 32 bf16)
  const int n0 = blockIdx.x * 128, m0 = blockIdx.y * 64;
  const int tid = threadIdx.x, lane = tid & 31, wave = tid >> 5;
  const int wm = wave & 1, wn = wave >> 1;          // 2 x 4 wave grid
  const int lane_lo = lane & 15, lane_hi = lane >> 4;

  v8f acc[2][2] = {};

#if USE_TDM
  // ---- Tensor Data Mover double-buffered pipeline ----
  const unsigned ldsA[2] = { (unsigned)(uintptr_t)&sA[0][0],
                             (unsigned)(uintptr_t)&sA[1][0] };
  const unsigned ldsB[2] = { (unsigned)(uintptr_t)&sB[0][0],
                             (unsigned)(uintptr_t)&sB[1][0] };
  if (wave == 0) {
    tdm_load_tile(ldsA[0], A + (size_t)m0 * K, 64, K);
    tdm_load_tile(ldsB[0], W + (size_t)n0 * K, 128, K);
  }
  for (int k0 = 0; k0 < K; k0 += 32) {
    const int cur = (k0 >> 5) & 1;
    wait_tensor0();          // issuing wave drains TENSORcnt before signaling
    __syncthreads();         // tiles in buf[cur] visible to all waves
    if (wave == 0 && k0 + 32 < K) {   // DMA next tiles while WMMAs run
      tdm_load_tile(ldsA[cur ^ 1], A + (size_t)m0 * K + (k0 + 32), 64, K);
      tdm_load_tile(ldsB[cur ^ 1], W + (size_t)n0 * K + (k0 + 32), 128, K);
    }
    const char* cA = sA[cur];
    const char* cB = sB[cur];
    v16bf af[2], bf[2];
#pragma unroll
    for (int i = 0; i < 2; i++)
      af[i] = frag_a(cA + (wm * 32 + i * 16 + lane_lo) * 64, lane_hi);
#pragma unroll
    for (int j = 0; j < 2; j++)
      bf[j] = frag_b(cB + (wn * 32 + j * 16 + lane_lo) * 64, lane_hi);
#pragma unroll
    for (int i = 0; i < 2; i++)
#pragma unroll
      for (int j = 0; j < 2; j++) acc[i][j] = wmma_bf16(af[i], bf[j], acc[i][j]);
  }
#else
  // ---- fallback: vector-load staging ----
  for (int k0 = 0; k0 < K; k0 += 32) {
    __syncthreads();
    {   // stage A tile 64x32
      int r = tid >> 2, c = (tid & 3) * 8;
      *(v4u*)(sA[0] + r * 64 + c * 2) = *(const v4u*)(A + (size_t)(m0 + r) * K + k0 + c);
    }
    {   // stage B tile 128x32 (n-major)
      int r = tid >> 1, c = (tid & 1) * 16;
      const unsigned short* wp = W + (size_t)(n0 + r) * K + k0 + c;
      *(v4u*)(sB[0] + r * 64 + c * 2)      = *(const v4u*)(wp);
      *(v4u*)(sB[0] + r * 64 + c * 2 + 16) = *(const v4u*)(wp + 8);
    }
    if (k0 + 32 < K) {
      __builtin_prefetch(A + (size_t)(m0 + (tid >> 2)) * K + k0 + 32 + (tid & 3) * 8, 0, 1);
      __builtin_prefetch(W + (size_t)(n0 + (tid >> 1)) * K + k0 + 32 + (tid & 1) * 16, 0, 1);
    }
    __syncthreads();
    v16bf af[2], bf[2];
#pragma unroll
    for (int i = 0; i < 2; i++)
      af[i] = frag_a(sA[0] + (wm * 32 + i * 16 + lane_lo) * 64, lane_hi);
#pragma unroll
    for (int j = 0; j < 2; j++)
      bf[j] = frag_b(sB[0] + (wn * 32 + j * 16 + lane_lo) * 64, lane_hi);
#pragma unroll
    for (int i = 0; i < 2; i++)
#pragma unroll
      for (int j = 0; j < 2; j++) acc[i][j] = wmma_bf16(af[i], bf[j], acc[i][j]);
  }
#endif

  // epilogue: C layout -> row = r + 8*lane_hi, col = lane_lo
#pragma unroll
  for (int i = 0; i < 2; i++)
#pragma unroll
    for (int j = 0; j < 2; j++) {
      int col = n0 + wn * 32 + j * 16 + lane_lo;
      float bv = bias[col];
#pragma unroll
      for (int r = 0; r < 8; r++) {
        int row = m0 + wm * 32 + i * 16 + r + 8 * lane_hi;
        float v = acc[i][j][r] + bv;
        if (MODE == 0) {
          outh[(size_t)row * N + col] = f2bf(v);
        } else if (MODE == 1) {
          float gl = 0.5f * v * (1.f + erff(v * 0.70710678118654752f));
          outh[(size_t)row * N + col] = f2bf(gl);
        } else if (MODE == 2) {
          size_t t = win2tok(row);
          outf[t * DIM + col] = v + res[t * DIM + col];
        } else {
          outf[(size_t)row * DIM + col] = v + res[(size_t)row * DIM + col];
        }
      }
    }
}

// ---------------- attention: one block per (window, head) ----------------
// qkv: bf16 [BWIN*NW*NTOK][768] (q|k|v, head-major 32-wide slices)
__global__ __launch_bounds__(256) void esb_attn(const unsigned short* __restrict__ qkv,
                                                const float* __restrict__ btab,
                                                unsigned short* __restrict__ winO) {
  extern __shared__ __align__(16) char smem[];
  float* Sf = (float*)smem;                               // [144][146] f32 scores
  char* Ps = smem + 144 * 146 * 4;                        // [144][160] bf16 probs
  char* Qs = Ps + 144 * 160 * 2;                          // [144][32]  bf16
  char* Ks = Qs + 144 * 32 * 2;                           // [144][32]  bf16
  char* Vt = Ks + 144 * 32 * 2;                           // [32][160]  bf16 (c-major)

  const int blk = blockIdx.x;
  const int h = blk & 7, win = blk >> 3, w = win % NW;
  const int tid = threadIdx.x, lane = tid & 31, wave = tid >> 5;
  const int lane_lo = lane & 15, lane_hi = lane >> 4;

  const unsigned short* base = qkv + (size_t)win * NTOK * 768 + h * HD;
  for (int idx = tid; idx < NTOK * HD; idx += 256) {
    int r = idx >> 5, c = idx & 31;
    const unsigned short* rp = base + (size_t)r * 768;
    ((unsigned short*)Qs)[r * 32 + c] = f2bf(bf2f(rp[c]) * 0.17677669529663687f);
    ((unsigned short*)Ks)[r * 32 + c] = rp[256 + c];
    ((unsigned short*)Vt)[c * 160 + r] = rp[512 + c];
  }
  for (int idx = tid; idx < 32 * 16; idx += 256)          // zero V pad rows (K 144..159)
    ((unsigned short*)Vt)[(idx >> 4) * 160 + 144 + (idx & 15)] = 0;
  __syncthreads();

  // S = Q*K^T + bias : 9x9 tiles, K=32 = one WMMA each
  for (int t = wave; t < 81; t += 8) {
    int qm = t / 9, kn = t % 9;
    v16bf a = frag_a(Qs + (qm * 16 + lane_lo) * 64, lane_hi);
    v16bf b = frag_b(Ks + (kn * 16 + lane_lo) * 64, lane_hi);
    v8f acc = {};
    acc = wmma_bf16(a, b, acc);
#pragma unroll
    for (int r = 0; r < 8; r++) {
      int qr = qm * 16 + r + 8 * lane_hi;
      int kc = kn * 16 + lane_lo;
      float bv = btab[((size_t)posidx(qr, kc) * NW + w) * HEADS + h];
      Sf[qr * 146 + kc] = acc[r] + bv;
    }
  }
  __syncthreads();

  // row softmax (fp32), emit bf16 P with zero pad columns
  for (int row = wave; row < NTOK; row += 8) {
    float* sr = Sf + row * 146;
    float mx = -3.4e38f;
    for (int j = lane; j < NTOK; j += 32) mx = fmaxf(mx, sr[j]);
#pragma unroll
    for (int o = 16; o; o >>= 1) mx = fmaxf(mx, __shfl_xor(mx, o, 32));
    float sum = 0.f;
    for (int j = lane; j < NTOK; j += 32) { float e = __expf(sr[j] - mx); sr[j] = e; sum += e; }
#pragma unroll
    for (int o = 16; o; o >>= 1) sum += __shfl_xor(sum, o, 32);
    float inv = 1.f / sum;
    unsigned short* pr = (unsigned short*)Ps + row * 160;
    for (int j = lane; j < 160; j += 32)
      pr[j] = (j < NTOK) ? f2bf(sr[j] * inv) : (unsigned short)0;
  }
  __syncthreads();

  // O = P*V : 9x2 tiles, K=160 -> 5 WMMA steps
  unsigned short* outp = winO + (size_t)win * NTOK * DIM + h * HD;
  for (int t = wave; t < 18; t += 8) {
    int qm = t >> 1, cn = t & 1;
    v8f acc = {};
#pragma unroll
    for (int kk = 0; kk < 5; kk++) {
      v16bf a = frag_a(Ps + (qm * 16 + lane_lo) * 320 + kk * 64, lane_hi);
      v16bf b = frag_b(Vt + (cn * 16 + lane_lo) * 320 + kk * 64, lane_hi);
      acc = wmma_bf16(a, b, acc);
    }
#pragma unroll
    for (int r = 0; r < 8; r++) {
      int qr = qm * 16 + r + 8 * lane_hi;
      outp[(size_t)qr * DIM + cn * 16 + lane_lo] = f2bf(acc[r]);
    }
  }
}

// ---------------- host orchestration ----------------
extern "C" void kernel_launch(void* const* d_in, const int* in_sizes, int n_in,
                              void* d_out, int out_size, void* d_ws, size_t ws_size,
                              hipStream_t stream) {
  const float* x      = (const float*)d_in[0];
  const float* n1g    = (const float*)d_in[1];
  const float* n1b    = (const float*)d_in[2];
  const float* qkv_w  = (const float*)d_in[3];
  const float* qkv_b  = (const float*)d_in[4];
  const float* btab   = (const float*)d_in[5];
  const float* proj_w = (const float*)d_in[6];
  const float* proj_b = (const float*)d_in[7];
  const float* n2g    = (const float*)d_in[8];
  const float* n2b    = (const float*)d_in[9];
  const float* fc1_w  = (const float*)d_in[10];
  const float* fc1_b  = (const float*)d_in[11];
  const float* fc2_w  = (const float*)d_in[12];
  const float* fc2_b  = (const float*)d_in[13];

  char* ws = (char*)d_ws;
  size_t o_winA = 0;
  size_t o_winO = o_winA + (size_t)MTOT * DIM * 2;
  size_t o_qkv  = o_winO + (size_t)MTOT * DIM * 2;     // winO+qkv contiguous (reused by fc1 out)
  size_t o_x2   = o_qkv  + (size_t)MTOT * 768 * 2;
  size_t o_wq   = o_x2   + (size_t)MTOT * DIM * 4;
  size_t o_wp   = o_wq   + (size_t)768 * 256 * 2;
  size_t o_w1   = o_wp   + (size_t)256 * 256 * 2;
  size_t o_w2   = o_w1   + (size_t)1024 * 256 * 2;

  unsigned short* winA = (unsigned short*)(ws + o_winA);
  unsigned short* winO = (unsigned short*)(ws + o_winO);
  unsigned short* qkvY = (unsigned short*)(ws + o_qkv);
  float*          x2   = (float*)(ws + o_x2);
  unsigned short* wq   = (unsigned short*)(ws + o_wq);
  unsigned short* wp   = (unsigned short*)(ws + o_wp);
  unsigned short* w1   = (unsigned short*)(ws + o_w1);
  unsigned short* w2   = (unsigned short*)(ws + o_w2);
  unsigned short* ln2o = winA;   // winA dead after QKV GEMM
  unsigned short* fc1o = winO;   // winO+qkvY dead after proj -> reuse region

  esb_cvt<<<(768 * 256 + 255) / 256, 256, 0, stream>>>(qkv_w, wq, 768 * 256);
  esb_cvt<<<(256 * 256 + 255) / 256, 256, 0, stream>>>(proj_w, wp, 256 * 256);
  esb_cvt<<<(1024 * 256 + 255) / 256, 256, 0, stream>>>(fc1_w, w1, 1024 * 256);
  esb_cvt<<<(256 * 1024 + 255) / 256, 256, 0, stream>>>(fc2_w, w2, 256 * 1024);

  esb_ln<<<MTOT / 8, 256, 0, stream>>>(x, n1g, n1b, winA, 1);

  esb_gemm<0><<<dim3(768 / 128, MTOT / 64), 256, 0, stream>>>(
      winA, wq, qkv_b, MTOT, 768, 256, nullptr, nullptr, qkvY);

  size_t smem_attn = 144 * 146 * 4 + 144 * 160 * 2 + 144 * 32 * 2 + 144 * 32 * 2 +
                     32 * 160 * 2;   // 158,848 B (<320KB/WG limit)
  (void)hipFuncSetAttribute((const void*)esb_attn,
                            hipFuncAttributeMaxDynamicSharedMemorySize, (int)smem_attn);
  esb_attn<<<BWIN * NW * HEADS, 256, smem_attn, stream>>>(qkvY, btab, winO);

  esb_gemm<2><<<dim3(256 / 128, MTOT / 64), 256, 0, stream>>>(
      winO, wp, proj_b, MTOT, 256, 256, x, x2, nullptr);

  esb_ln<<<MTOT / 8, 256, 0, stream>>>(x2, n2g, n2b, ln2o, 0);

  esb_gemm<1><<<dim3(1024 / 128, MTOT / 64), 256, 0, stream>>>(
      ln2o, w1, fc1_b, MTOT, 1024, 256, nullptr, nullptr, fc1o);

  esb_gemm<3><<<dim3(256 / 128, MTOT / 64), 256, 0, stream>>>(
      fc1o, w2, fc2_b, MTOT, 256, 1024, x2, (float*)d_out, nullptr);
}
